// Attention_13271448945073
// MI455X (gfx1250) — compile-verified, matrix operand order checked by reading, not run on hardware
//
#include <hip/hip_runtime.h>
#include <stdint.h>

#define B_SZ   2
#define NSEQ   2048
#define DMODEL 1024
#define HEADS  16
#define DHEAD  64
#define MROWS  (B_SZ * NSEQ)   // 4096 token rows

typedef __attribute__((ext_vector_type(16))) __bf16        v16bf;
typedef __attribute__((ext_vector_type(8)))  float         v8f;
typedef __attribute__((ext_vector_type(4)))  unsigned int  v4u;
typedef __attribute__((ext_vector_type(4)))  float         v4f;
typedef __attribute__((ext_vector_type(4)))  unsigned short v4us;
typedef unsigned short u16;

union Frag { v16bf v; v4u q[2]; };

__device__ __forceinline__ u16 bf16_rne(float f) {
  unsigned int u = __float_as_uint(f);
  u += 0x7fffu + ((u >> 16) & 1u);   // round-to-nearest-even
  return (u16)(u >> 16);
}

// Load a 16x32 bf16 WMMA fragment (A layout; B layout by symmetry over columns).
// Lane l<16 : row l, K {0..7, 16..23}; lane l+16: row l, K {8..15, 24..31}.
// Works for both global and LDS pointers (addrspace inferred).
__device__ __forceinline__ v16bf load_frag16x32(const u16* base, int stride) {
  const int lane = threadIdx.x & 31;
  const u16* p = base + (long)(lane & 15) * stride + ((lane >> 4) << 3);
  Frag f;
  f.q[0] = *(const v4u*)(p);
  f.q[1] = *(const v4u*)(p + 16);
  return f.v;
}

__device__ __forceinline__ v8f wmma_bf16(v16bf a, v16bf b, v8f c) {
  // (neg_a, A, neg_b, B, c_mod, C, reuse_a, reuse_b)
  return __builtin_amdgcn_wmma_f32_16x16x32_bf16(false, a, false, b, (short)0, c, false, false);
}

// CDNA5 async global->LDS copy, 16B per lane, tracked with ASYNCcnt.
__device__ __forceinline__ void async_ld_b128(uint32_t lds_off, const u16* gptr) {
  asm volatile("global_load_async_to_lds_b128 %0, %1, off"
               :: "v"(lds_off), "v"(gptr) : "memory");
}

// ---------------------------------------------------------------------------
// f32 -> bf16 conversion (bandwidth trivial)
// ---------------------------------------------------------------------------
__global__ void cvt_bf16_kernel(const float* __restrict__ in, u16* __restrict__ out, int n4) {
  int i = blockIdx.x * blockDim.x + threadIdx.x;
  if (i < n4) {
    v4f f = ((const v4f*)in)[i];
    v4us o;
    o.x = bf16_rne(f.x); o.y = bf16_rne(f.y);
    o.z = bf16_rne(f.z); o.w = bf16_rne(f.w);
    ((v4us*)out)[i] = o;
  }
}

// ---------------------------------------------------------------------------
// C = A(MxK, bf16 row-major) x W^T (W is Nc x K, bf16 row-major)
// 8 waves/block; wave = 16 rows x 64 cols; block tile = 128 x 64.
// Double-buffered async global->LDS staging; rows padded for conflict-free
// ds_load_b128 fragment reads.
// mode 0: bf16 row-major.  mode 1: bf16 as V^T (b,h,d,n).
// mode 2: f32 + bias + residual (final projection).
// ---------------------------------------------------------------------------
#define LDSROW 40   // 32 elems + 8 pad

__global__ __launch_bounds__(256)
void gemm_wmma_kernel(const u16* __restrict__ A,
                      const u16* __restrict__ Bw,
                      u16* __restrict__ Obf,
                      float* __restrict__ Of32,
                      const float* __restrict__ bias,
                      const float* __restrict__ residual,
                      int K, int Nc, int mode)
{
  __shared__ __align__(16) u16 sA[2][128 * LDSROW];  // 2 x 10240 B
  __shared__ __align__(16) u16 sB[2][64 * LDSROW];   // 2 x  5120 B

  const int tid  = threadIdx.x;
  const int wave = tid >> 5;
  const int lane = tid & 31;
  const int m0 = blockIdx.x * 128 + wave * 16;
  const int n0 = blockIdx.y * 64;

  // --- cooperative staging assignments: 16B chunks ---
  const int ar0 = tid >> 2,          ac0 = tid & 3;     // A: 128 rows x 4 chunks (x2)
  const int ar1 = (tid + 256) >> 2;                     // ac1 == ac0
  const int br  = tid >> 2,          bc  = tid & 3;     // B: 64 rows x 4 chunks

  const u16* gA0 = A  + (long)(blockIdx.x * 128 + ar0) * K + ac0 * 8;
  const u16* gA1 = A  + (long)(blockIdx.x * 128 + ar1) * K + ac0 * 8;
  const u16* gB  = Bw + (long)(n0 + br) * K + bc * 8;

  uint32_t lA0[2], lA1[2], lB[2];
#pragma unroll
  for (int buf = 0; buf < 2; ++buf) {
    lA0[buf] = (uint32_t)(uintptr_t)&sA[buf][ar0 * LDSROW + ac0 * 8];
    lA1[buf] = (uint32_t)(uintptr_t)&sA[buf][ar1 * LDSROW + ac0 * 8];
    lB [buf] = (uint32_t)(uintptr_t)&sB[buf][br  * LDSROW + bc  * 8];
  }

  v8f acc[4] = {};
  const int KSTEPS = K >> 5;

  // prologue: stage k-step 0 into buffer 0
  async_ld_b128(lA0[0], gA0);
  async_ld_b128(lA1[0], gA1);
  async_ld_b128(lB [0], gB);

  for (int ks = 0; ks < KSTEPS; ++ks) {
    const int cur = ks & 1;
    if (ks + 1 < KSTEPS) {
      const int k0n = (ks + 1) << 5;
      async_ld_b128(lA0[cur ^ 1], gA0 + k0n);
      async_ld_b128(lA1[cur ^ 1], gA1 + k0n);
      async_ld_b128(lB [cur ^ 1], gB  + k0n);
      // async loads retire in order: older group done when <=3 remain
      asm volatile("s_wait_asynccnt 0x3" ::: "memory");
    } else {
      asm volatile("s_wait_asynccnt 0x0" ::: "memory");
    }
    __syncthreads();

    const u16* pA = &sA[cur][(wave * 16 + (lane & 15)) * LDSROW + ((lane >> 4) << 3)];
    Frag fa;
    fa.q[0] = *(const v4u*)(pA);
    fa.q[1] = *(const v4u*)(pA + 16);
#pragma unroll
    for (int f = 0; f < 4; ++f) {
      const u16* pB = &sB[cur][(16 * f + (lane & 15)) * LDSROW + ((lane >> 4) << 3)];
      Frag fb;
      fb.q[0] = *(const v4u*)(pB);
      fb.q[1] = *(const v4u*)(pB + 16);
      acc[f] = wmma_bf16(fa.v, fb.v, acc[f]);
    }
    __syncthreads();   // all waves done reading before buffer is overwritten
  }

  // --- epilogue ---
  const int col = lane & 15;
  const int rb  = (lane >> 4) << 3;   // C layout: row = vgpr + 8*(lane>=16)
  if (mode == 0) {
#pragma unroll
    for (int f = 0; f < 4; ++f) {
      int n = n0 + 16 * f + col;
#pragma unroll
      for (int r = 0; r < 8; ++r)
        Obf[(long)(m0 + rb + r) * Nc + n] = bf16_rne(acc[f][r]);
    }
  } else if (mode == 1) {
#pragma unroll
    for (int f = 0; f < 4; ++f) {
      int n = n0 + 16 * f + col;
      int h = n >> 6, d = n & 63;
#pragma unroll
      for (int r = 0; r < 8; ++r) {
        int m = m0 + rb + r;
        int bb = m >> 11, tok = m & (NSEQ - 1);
        Obf[(((long)bb * HEADS + h) * DHEAD + d) * NSEQ + tok] = bf16_rne(acc[f][r]);
      }
    }
  } else {
#pragma unroll
    for (int f = 0; f < 4; ++f) {
      int n = n0 + 16 * f + col;
      float bn = bias[n];
#pragma unroll
      for (int r = 0; r < 8; ++r) {
        long idx = (long)(m0 + rb + r) * Nc + n;
        Of32[idx] = acc[f][r] + bn + residual[idx];
      }
    }
  }
}

// ---------------------------------------------------------------------------
// Flash attention: block = 128 query rows of one (b,h); wave owns 16 rows.
// K/V tiles for the current 32-key step are shared by all 8 waves, so they
// are staged once per block via double-buffered async global->LDS copies.
// Q,K: (B,N,H*64) bf16.  VT: (B,H,64,N) bf16.  O: (B,N,H*64) bf16.
// ---------------------------------------------------------------------------
#define KROW 72   // 64 elems + 8 pad (144B: 9*l mod 16 -> conflict-free)
#define VROW 40   // 32 elems + 8 pad

__global__ __launch_bounds__(256)
void flash_attn_kernel(const u16* __restrict__ Q,
                       const u16* __restrict__ Kmat,
                       const u16* __restrict__ VT,
                       u16* __restrict__ O)
{
  __shared__ __align__(16) u16 sK[2][32 * KROW];   // 2 x 4608 B (32 keys x 64 d)
  __shared__ __align__(16) u16 sV[2][64 * VROW];   // 2 x 5120 B (64 d x 32 keys)
  __shared__ __align__(16) u16 ldsP[8][16 * 32];   // per-wave P staging (8 KB)

  const int tid  = threadIdx.x;
  const int wave = tid >> 5;
  const int lane = tid & 31;
  const int b = blockIdx.y / HEADS;
  const int h = blockIdx.y % HEADS;
  const int q0 = blockIdx.x * 128 + wave * 16;
  const int tokStride = HEADS * DHEAD;  // 1024

  // --- staging assignments: 16B chunks, 1 K-chunk + 1 V-chunk per thread ---
  const int krow = tid >> 3, kcn = tid & 7;   // K tile: 32 rows x 8 chunks
  const int vrow = tid >> 2, vcn = tid & 3;   // V tile: 64 rows x 4 chunks
  const u16* gK = Kmat + (((long)b * NSEQ + krow) * HEADS + h) * DHEAD + kcn * 8;
  const u16* gV = VT   + (((long)b * HEADS + h) * DHEAD + vrow) * NSEQ + vcn * 8;
  uint32_t lK[2], lV[2];
#pragma unroll
  for (int buf = 0; buf < 2; ++buf) {
    lK[buf] = (uint32_t)(uintptr_t)&sK[buf][krow * KROW + kcn * 8];
    lV[buf] = (uint32_t)(uintptr_t)&sV[buf][vrow * VROW + vcn * 8];
  }

  // --- Q fragments: resident for the whole kernel ---
  const u16* Qbase = Q + (((long)b * NSEQ + q0) * HEADS + h) * DHEAD;
  v16bf aQ0 = load_frag16x32(Qbase, tokStride);        // d 0..31
  v16bf aQ1 = load_frag16x32(Qbase + 32, tokStride);   // d 32..63

  v8f o[4] = {};
  float mrow[8], lrow[8];
#pragma unroll
  for (int r = 0; r < 8; ++r) { mrow[r] = -1e30f; lrow[r] = 0.f; }
  const float scale = 0.125f;   // 1/sqrt(64)

  const int JSTEPS = NSEQ / 32;

  // prologue: stage key tile 0 into buffer 0
  async_ld_b128(lK[0], gK);
  async_ld_b128(lV[0], gV);

  for (int js = 0; js < JSTEPS; ++js) {
    const int cur = js & 1;
    if (js + 1 < JSTEPS) {
      const int j0n = (js + 1) << 5;
      async_ld_b128(lK[cur ^ 1], gK + (long)j0n * tokStride);
      async_ld_b128(lV[cur ^ 1], gV + j0n);
      asm volatile("s_wait_asynccnt 0x2" ::: "memory");
    } else {
      asm volatile("s_wait_asynccnt 0x0" ::: "memory");
    }
    __syncthreads();

    // ---- S = Q K^T (16 x 32 tile = two 16x16 C fragments) ----
    v8f s0 = {}, s1 = {};
    {
      const u16* kb = &sK[cur][0];
      v16bf b0 = load_frag16x32(kb, KROW);                  // keys 0..15, d 0..31
      v16bf b1 = load_frag16x32(kb + 16 * KROW, KROW);      // keys 16..31
      s0 = wmma_bf16(aQ0, b0, s0);
      s1 = wmma_bf16(aQ0, b1, s1);
      b0 = load_frag16x32(kb + 32, KROW);                   // d 32..63
      b1 = load_frag16x32(kb + 16 * KROW + 32, KROW);
      s0 = wmma_bf16(aQ1, b0, s0);
      s1 = wmma_bf16(aQ1, b1, s1);
    }

    // ---- online softmax: row stats via 16-lane shuffle reductions ----
    float alpha[8];
#pragma unroll
    for (int r = 0; r < 8; ++r) {
      float a0 = s0[r] * scale;
      float a1 = s1[r] * scale;
      float mx = fmaxf(a0, a1);
#pragma unroll
      for (int off = 1; off < 16; off <<= 1)
        mx = fmaxf(mx, __shfl_xor(mx, off, 32));
      float mnew = fmaxf(mrow[r], mx);
      alpha[r] = __expf(mrow[r] - mnew);
      float p0 = __expf(a0 - mnew);
      float p1 = __expf(a1 - mnew);
      s0[r] = p0; s1[r] = p1;
      float ps = p0 + p1;
#pragma unroll
      for (int off = 1; off < 16; off <<= 1)
        ps += __shfl_xor(ps, off, 32);
      lrow[r] = lrow[r] * alpha[r] + ps;
      mrow[r] = mnew;
    }
#pragma unroll
    for (int f = 0; f < 4; ++f)
#pragma unroll
      for (int r = 0; r < 8; ++r)
        o[f][r] *= alpha[r];

    // ---- P: C-layout f32 -> row-major bf16 in LDS -> A fragment ----
    {
      u16* pl = &ldsP[wave][0];
      const int rbw = (lane >> 4) << 3;
      const int cw  = lane & 15;
#pragma unroll
      for (int r = 0; r < 8; ++r) {
        pl[(rbw + r) * 32 + cw]      = bf16_rne(s0[r]);
        pl[(rbw + r) * 32 + 16 + cw] = bf16_rne(s1[r]);
      }
    }
    __builtin_amdgcn_wave_barrier();
    v16bf aP;
    {
      Frag f;
      const u16* pl = &ldsP[wave][0] + (lane & 15) * 32 + ((lane >> 4) << 3);
      f.q[0] = *(const v4u*)(pl);
      f.q[1] = *(const v4u*)(pl + 16);
      aP = f.v;
    }
    __builtin_amdgcn_wave_barrier();

    // ---- O += P * V from staged V^T tile ----
#pragma unroll
    for (int f = 0; f < 4; ++f) {
      v16bf bV = load_frag16x32(&sV[cur][(16 * f) * VROW], VROW);
      o[f] = wmma_bf16(aP, bV, o[f]);
    }
    __syncthreads();   // all waves done with this buffer before it is refilled
  }

  // ---- finalize: O /= l, store bf16 (B,N,H*64) ----
  const int col = lane & 15;
  const int rb  = (lane >> 4) << 3;
  float inv[8];
#pragma unroll
  for (int r = 0; r < 8; ++r) inv[r] = 1.0f / lrow[r];
#pragma unroll
  for (int f = 0; f < 4; ++f) {
#pragma unroll
    for (int r = 0; r < 8; ++r) {
      long idx = (((long)b * NSEQ + q0 + rb + r) * HEADS + h) * DHEAD + 16 * f + col;
      O[idx] = bf16_rne(o[f][r] * inv[r]);
    }
  }
}

// ---------------------------------------------------------------------------
// Host launcher
// ---------------------------------------------------------------------------
extern "C" void kernel_launch(void* const* d_in, const int* in_sizes, int n_in,
                              void* d_out, int out_size, void* d_ws, size_t ws_size,
                              hipStream_t stream)
{
  const float* q_f = (const float*)d_in[0];
  const float* k_f = (const float*)d_in[1];
  const float* v_f = (const float*)d_in[2];
  const float* wq  = (const float*)d_in[3];
  const float* wk  = (const float*)d_in[4];
  const float* wv  = (const float*)d_in[5];
  const float* wo  = (const float*)d_in[6];
  const float* bo  = (const float*)d_in[7];

  const long TOK = (long)MROWS * DMODEL;    // 4,194,304 elems
  const long WEL = (long)DMODEL * DMODEL;   // 1,048,576 elems

  u16* ws  = (u16*)d_ws;                    // needs 64 MB total
  u16* Xq  = ws;
  u16* Xk  = Xq  + TOK;
  u16* Xv  = Xk  + TOK;
  u16* Wqb = Xv  + TOK;
  u16* Wkb = Wqb + WEL;
  u16* Wvb = Wkb + WEL;
  u16* Wob = Wvb + WEL;
  u16* Qb  = Wob + WEL;
  u16* Kb  = Qb  + TOK;
  u16* VTb = Kb  + TOK;
  u16* Ab  = VTb + TOK;

  // f32 -> bf16
  {
    int n4 = (int)(TOK >> 2);
    int blk = 256, grd = (n4 + blk - 1) / blk;
    cvt_bf16_kernel<<<grd, blk, 0, stream>>>(q_f, Xq, n4);
    cvt_bf16_kernel<<<grd, blk, 0, stream>>>(k_f, Xk, n4);
    cvt_bf16_kernel<<<grd, blk, 0, stream>>>(v_f, Xv, n4);
    int w4 = (int)(WEL >> 2), wgrd = (w4 + blk - 1) / blk;
    cvt_bf16_kernel<<<wgrd, blk, 0, stream>>>(wq, Wqb, w4);
    cvt_bf16_kernel<<<wgrd, blk, 0, stream>>>(wk, Wkb, w4);
    cvt_bf16_kernel<<<wgrd, blk, 0, stream>>>(wv, Wvb, w4);
    cvt_bf16_kernel<<<wgrd, blk, 0, stream>>>(wo, Wob, w4);
  }

  // Projections (Q, K row-major; V stored transposed for the PV GEMM)
  dim3 gg(MROWS / 128, DMODEL / 64);
  gemm_wmma_kernel<<<gg, 256, 0, stream>>>(Xq, Wqb, Qb,  nullptr, nullptr, nullptr, DMODEL, DMODEL, 0);
  gemm_wmma_kernel<<<gg, 256, 0, stream>>>(Xk, Wkb, Kb,  nullptr, nullptr, nullptr, DMODEL, DMODEL, 0);
  gemm_wmma_kernel<<<gg, 256, 0, stream>>>(Xv, Wvb, VTb, nullptr, nullptr, nullptr, DMODEL, DMODEL, 1);

  // Flash attention
  dim3 ga(NSEQ / 128, B_SZ * HEADS);
  flash_attn_kernel<<<ga, 256, 0, stream>>>(Qb, Kb, VTb, Ab);

  // Output projection + bias + residual, f32 out
  gemm_wmma_kernel<<<gg, 256, 0, stream>>>(Ab, Wob, nullptr, (float*)d_out, bo, q_f, DMODEL, DMODEL, 2);

  (void)in_sizes; (void)n_in; (void)out_size; (void)ws_size;
}
